// WindowAttention_51221779972613
// MI455X (gfx1250) — matrix operand order repaired and checked by reading, not compiled, not run on hardware
//
#include <hip/hip_runtime.h>

#define WIN 256
#define DIM 256

typedef __bf16 bf16_t;
typedef __attribute__((ext_vector_type(16))) __bf16 v16bf;
typedef __attribute__((ext_vector_type(2))) __bf16 v2bf;
typedef __attribute__((ext_vector_type(8))) float v8f;

union Frag {
  v16bf bf;
  unsigned u[8];
  uint4 q[2];
};

// Pack two f32 -> packed bf16 (round-to-nearest-even).
__device__ __forceinline__ unsigned pk_bf16(float lo, float hi) {
#if __has_builtin(__builtin_amdgcn_cvt_pk_bf16_f32)
  v2bf p = __builtin_amdgcn_cvt_pk_bf16_f32(lo, hi);
  return __builtin_bit_cast(unsigned, p);
#else
  unsigned a = __builtin_bit_cast(unsigned, lo);
  unsigned b = __builtin_bit_cast(unsigned, hi);
  a = a + 0x7fffu + ((a >> 16) & 1u);
  b = b + 0x7fffu + ((b >> 16) & 1u);
  return (a >> 16) | (b & 0xffff0000u);
#endif
}

__device__ __forceinline__ bf16_t cvt_bf16(float x) {
#if __has_builtin(__builtin_amdgcn_cvt_pk_bf16_f32)
  v2bf p = __builtin_amdgcn_cvt_pk_bf16_f32(x, x);
  return p.x;
#else
  unsigned a = __builtin_bit_cast(unsigned, x);
  a = a + 0x7fffu + ((a >> 16) & 1u);
  unsigned short s = (unsigned short)(a >> 16);
  return __builtin_bit_cast(bf16_t, s);
#endif
}

// Bitmask swizzle across the wave32: group-of-32, and=0x10, or=0, xor=0x1f (reads lane (l & 16) ^ 31).
__device__ __forceinline__ float xor16(float x) {
  int i = __builtin_bit_cast(int, x);
  i = __builtin_amdgcn_ds_swizzle(i, 0x7C10);
  return __builtin_bit_cast(float, i);
}

__global__ __launch_bounds__(256, 1) void win_attn_kernel(
    const float* __restrict__ qg_, const float* __restrict__ kg_,
    const float* __restrict__ vg_, float* __restrict__ og_) {
  extern __shared__ char smem[];
  bf16_t* kbuf = (bf16_t*)smem;      // [key][chan] bf16, 128 KB
  bf16_t* vtbuf = kbuf + WIN * DIM;  // [chan][key] bf16, 128 KB

  const int tid = threadIdx.x;
  const size_t wbase = (size_t)blockIdx.x * (WIN * DIM);
  const float* qg = qg_ + wbase;
  const float* kg = kg_ + wbase;
  const float* vg = vg_ + wbase;
  float* og = og_ + wbase;

  // ---- Stage K (row-major) and V^T into LDS as bf16 (read HBM once). ----
  #pragma unroll 2
  for (int it = 0; it < 64; ++it) {
    int e = it * 1024 + tid * 4;  // flat element index, float4 granularity
    float4 kv = *(const float4*)(kg + e);
    uint2 pk = make_uint2(pk_bf16(kv.x, kv.y), pk_bf16(kv.z, kv.w));
    *(uint2*)(kbuf + e) = pk;
    float4 vv = *(const float4*)(vg + e);
    int kk = e >> 8;   // key row
    int c0 = e & 255;  // channel
    vtbuf[(c0 + 0) * WIN + kk] = cvt_bf16(vv.x);
    vtbuf[(c0 + 1) * WIN + kk] = cvt_bf16(vv.y);
    vtbuf[(c0 + 2) * WIN + kk] = cvt_bf16(vv.z);
    vtbuf[(c0 + 3) * WIN + kk] = cvt_bf16(vv.w);
  }
  __syncthreads();

  const int wave = tid >> 5;
  const int lane = tid & 31;
  const int lg = lane >> 4;  // half-wave group (ISA A/B layout "g")
  const int ln = lane & 15;

  const float scale = 0.0625f;  // DIM^-0.5 = 1/16, exact in bf16

  for (int rt = 0; rt < 2; ++rt) {
    const int q0 = wave * 32 + rt * 16;  // 16-query row tile

    // ---- GEMM1: S^T[key][query] = K @ Q^T. kt OUTER and rolled so only one
    // Q fragment (8 VGPRs) is live at a time; s[16] accumulators stay live.
    v8f s[16];
    #pragma unroll
    for (int tn = 0; tn < 16; ++tn)
      s[tn] = (v8f){0.f, 0.f, 0.f, 0.f, 0.f, 0.f, 0.f, 0.f};

    const float* qrow = qg + (q0 + ln) * DIM;
    // Software-pipeline the raw Q float4s across kt so each iteration's
    // global loads are issued one full WMMA sequence ahead of their use.
    float4 t0, t1, t2, t3;
    {
      const float* qp = qrow + lg * 16;
      t0 = *(const float4*)(qp + 0);
      t1 = *(const float4*)(qp + 4);
      t2 = *(const float4*)(qp + 8);
      t3 = *(const float4*)(qp + 12);
    }
    #pragma clang loop unroll(disable)
    for (int kt = 0; kt < 8; ++kt) {
      // Prefetch next kt's raw Q (branchless wrap; kt=7 reloads kt=0, unused).
      float4 n0, n1, n2, n3;
      {
        const float* qp = qrow + ((kt + 1) & 7) * 32 + lg * 16;
        n0 = *(const float4*)(qp + 0);
        n1 = *(const float4*)(qp + 4);
        n2 = *(const float4*)(qp + 8);
        n3 = *(const float4*)(qp + 12);
      }
      // Q^T fragment (B operand). B 32x16 layout: lane column m=ln,
      // VGPR j <- channels 32kt + 16g + 2j,2j+1.
      Frag qf;
      qf.u[0] = pk_bf16(t0.x * scale, t0.y * scale);
      qf.u[1] = pk_bf16(t0.z * scale, t0.w * scale);
      qf.u[2] = pk_bf16(t1.x * scale, t1.y * scale);
      qf.u[3] = pk_bf16(t1.z * scale, t1.w * scale);
      qf.u[4] = pk_bf16(t2.x * scale, t2.y * scale);
      qf.u[5] = pk_bf16(t2.z * scale, t2.w * scale);
      qf.u[6] = pk_bf16(t3.x * scale, t3.y * scale);
      qf.u[7] = pk_bf16(t3.z * scale, t3.w * scale);
      // A = K tile from LDS: lane row n=16tn+ln,
      // VGPR j <- chan 32kt + 16g + 2j,2j+1
      // => two contiguous ds_load_b128 per fragment; scheduler issues tile
      // tn+1's loads behind wmma(tn), overlapping LDS latency with the XDL op.
      #pragma unroll
      for (int tn = 0; tn < 16; ++tn) {
        const bf16_t* p = kbuf + (tn * 16 + ln) * DIM + kt * 32;
        Frag af;
        af.q[0] = *(const uint4*)(p + lg * 16);
        af.q[1] = *(const uint4*)(p + lg * 16 + 8);
        s[tn] = __builtin_amdgcn_wmma_f32_16x16x32_bf16(
            false, af.bf, false, qf.bf, (short)0, s[tn], false, false);
      }
      t0 = n0; t1 = n1; t2 = n2; t3 = n3;
    }

    // ---- Softmax over keys. In S^T D-layout, lane l holds query m=ln with
    // key offsets {r + 8*lg} per tile; other key half lives in lane^16.
    float m = -3.402823466e38f;
    #pragma unroll
    for (int tn = 0; tn < 16; ++tn)
      #pragma unroll
      for (int r = 0; r < 8; ++r) m = fmaxf(m, s[tn][r]);
    m = fmaxf(m, xor16(m));
    float sum = 0.f;
    #pragma unroll
    for (int tn = 0; tn < 16; ++tn)
      #pragma unroll
      for (int r = 0; r < 8; ++r) {
        float e = __expf(s[tn][r] - m);
        s[tn][r] = e;
        sum += e;
      }
    sum += xor16(sum);
    float inv = 1.f / sum;

    // ---- Repack P as bf16 A fragments for O = P @ V. The S^T D-layout maps
    // in-lane onto the 16-bit A 16x32 layout:
    //   A frag kt, dword j  <-  s[2kt + (j>=4)][2*(j%4) .. +1]   (same lane)
    // Built in kt order so s tiles die progressively as pf grows.
    Frag pf[8];
    #pragma unroll
    for (int kt = 0; kt < 8; ++kt)
      #pragma unroll
      for (int j = 0; j < 8; ++j) {
        int tn = kt * 2 + (j >> 2);
        int r = (j & 3) * 2;
        pf[kt].u[j] = pk_bf16(s[tn][r] * inv, s[tn][r + 1] * inv);
      }

    // ---- GEMM2: O = P @ V. B = V block (keys x chans) from V^T LDS:
    // lane column c=16tc+ln, VGPR j <- keys 32kt + (j<4 ? 2j+8g : 16+2(j-4)+8g)
    // => two contiguous ds_load_b128 per fragment.
    #pragma clang loop unroll(disable)
    for (int tc = 0; tc < 16; ++tc) {
      v8f acc = {0.f, 0.f, 0.f, 0.f, 0.f, 0.f, 0.f, 0.f};
      const bf16_t* vrow = vtbuf + (tc * 16 + ln) * WIN;
      #pragma unroll
      for (int kt = 0; kt < 8; ++kt) {
        const bf16_t* p = vrow + kt * 32 + lg * 8;
        Frag bfr;
        bfr.q[0] = *(const uint4*)(p);
        bfr.q[1] = *(const uint4*)(p + 16);
        acc = __builtin_amdgcn_wmma_f32_16x16x32_bf16(
            false, pf[kt].bf, false, bfr.bf, (short)0, acc, false, false);
      }
      // D layout: row q = q0 + r + 8*lg, col c = 16*tc + ln (coalesced f32).
      #pragma unroll
      for (int r = 0; r < 8; ++r)
        og[(q0 + r + 8 * lg) * DIM + tc * 16 + ln] = acc[r];
    }
  }
}

extern "C" void kernel_launch(void* const* d_in, const int* in_sizes, int n_in,
                              void* d_out, int out_size, void* d_ws, size_t ws_size,
                              hipStream_t stream) {
  const float* q = (const float*)d_in[0];
  const float* k = (const float*)d_in[1];
  const float* v = (const float*)d_in[2];
  float* out = (float*)d_out;
  // One workgroup per (batch, window): 16 * (8192/256) = 512 windows.
  int nwin = in_sizes[0] / (WIN * DIM);
  size_t smem_bytes = 2u * WIN * DIM * sizeof(bf16_t);  // 256 KB (K + V^T)
  win_attn_kernel<<<nwin, 256, smem_bytes, stream>>>(q, k, v, out);
}
